// MultiHeadDecoderPDTSP_7739531067806
// MI455X (gfx1250) — matrix operand order repaired and checked by reading, not compiled, not run on hardware
//
#include <hip/hip_runtime.h>
#include <hip/hip_bf16.h>
#include <math.h>

typedef __attribute__((ext_vector_type(16))) _Float16 v16h;
typedef __attribute__((ext_vector_type(8)))  float    v8f;

#define BS   128
#define GS   201
#define DIM  128
#define HALF 100
#define MROWS (BS*GS)        // 25728, divisible by 16
#define GG   (GS*GS)         // 40401
#define OUTW (HALF + GG)     // 40501

__device__ inline v8f wmma16x16x32(v16h a, v16h b, v8f c) {
  return __builtin_amdgcn_wmma_f32_16x16x32_f16(false, a, false, b, (short)0, c, false, false);
}

// ---------------- small prep kernels ----------------

__global__ void gmax_kernel(const float* __restrict__ h_em, float* __restrict__ gmax) {
  int b = blockIdx.x, d = threadIdx.x;
  const float* base = h_em + (size_t)b * GS * DIM + d;
  float mx = -INFINITY;
  for (int g = 0; g < GS; ++g) mx = fmaxf(mx, base[(size_t)g * DIM]);
  gmax[b * DIM + d] = mx;
}

__global__ void gproj_kernel(const float* __restrict__ gmax, const float* __restrict__ Wg,
                             float* __restrict__ gproj) {
  int b = blockIdx.x, d = threadIdx.x;
  float s = 0.f;
  for (int k = 0; k < DIM; ++k) s += gmax[b * DIM + k] * Wg[k * DIM + d];
  gproj[b * DIM + d] = s;
}

// Pack a K=128 x N=128 f32 weight into f16 WMMA B-fragment layout:
//   P[((nt*4 + kch)*32 + lane)*16 + t] = B[kch*32 + hi*8 + (t&7) + (t>>3)*16][nt*16 + l2]
// mode 0: B row-major [128][128];  mode 1: B[k][n] = WQr[n/32][k][n%32]
__global__ void pack_bfrag_kernel(const float* __restrict__ B, _Float16* __restrict__ P, int mode) {
  int idx = blockIdx.x * 256 + threadIdx.x;   // 8 nt * 4 kch * 32 lanes * 16 t = 16384
  if (idx >= 16384) return;
  int t    = idx & 15;
  int lane = (idx >> 4) & 31;
  int kch  = (idx >> 9) & 3;
  int nt   = idx >> 11;
  int l2 = lane & 15, hi = lane >> 4;
  int k = kch * 32 + hi * 8 + (t & 7) + (t >> 3) * 16;
  int n = nt * 16 + l2;
  float v = (mode == 0) ? B[k * DIM + n]
                        : B[(n >> 5) * (DIM * 32) + k * 32 + (n & 31)];
  P[idx] = (_Float16)v;
}

// Pack i_w2 [32][32] into per-lane f16 fragments:
//   P[lane*32 + which*16 + t] = w2[(hi*8 + (t&7) + (t>>3)*16)*32 + which*16 + l2]
__global__ void pack_w2_kernel(const float* __restrict__ w2, _Float16* __restrict__ P) {
  for (int i = threadIdx.x; i < 1024; i += blockDim.x) {
    int t = i & 15;
    int which = (i >> 4) & 1;
    int lane = i >> 5;
    int l2 = lane & 15, hi = lane >> 4;
    P[i] = (_Float16)w2[(hi * 8 + (t & 7) + (t >> 3) * 16) * 32 + which * 16 + l2];
  }
}

// ---------------- WMMA GEMM: D[M,128] = A[M,128] @ B (+ rowwise addvec[b][n]) ---------
// one wave per block; grid = (M/16, 8); Bpk pre-packed into fragment layout
__global__ void gemm_wmma(const float* __restrict__ A, const _Float16* __restrict__ Bpk,
                          float* __restrict__ D, const float* __restrict__ addvec) {
  constexpr int N = DIM;
  const int lane = threadIdx.x;
  const int l2 = lane & 15, hi = lane >> 4;
  const int m0 = blockIdx.x * 16, n0 = blockIdx.y * 16;
  const int m = m0 + l2, n = n0 + l2;
  const _Float16* bptr = Bpk + ((size_t)(blockIdx.y * 4) * 32 + lane) * 16;
  const float* arowb = A + (size_t)m * DIM + hi * 8;
  v8f c = {};
#pragma unroll
  for (int kc = 0; kc < 4; ++kc) {
    const float* arow = arowb + kc * 32;
    float4 a0 = *(const float4*)(arow);
    float4 a1 = *(const float4*)(arow + 4);
    float4 a2 = *(const float4*)(arow + 16);
    float4 a3 = *(const float4*)(arow + 20);
    v16h a;
    a[0]=(_Float16)a0.x;  a[1]=(_Float16)a0.y;  a[2]=(_Float16)a0.z;  a[3]=(_Float16)a0.w;
    a[4]=(_Float16)a1.x;  a[5]=(_Float16)a1.y;  a[6]=(_Float16)a1.z;  a[7]=(_Float16)a1.w;
    a[8]=(_Float16)a2.x;  a[9]=(_Float16)a2.y;  a[10]=(_Float16)a2.z; a[11]=(_Float16)a2.w;
    a[12]=(_Float16)a3.x; a[13]=(_Float16)a3.y; a[14]=(_Float16)a3.z; a[15]=(_Float16)a3.w;
    v16h b = *(const v16h*)(bptr + (size_t)kc * 32 * 16);
    c = wmma16x16x32(a, b, c);
  }
  if (addvec) {
#pragma unroll
    for (int r = 0; r < 8; ++r) {
      int mm = m0 + r + hi * 8;
      int bb = mm / GS;
      D[(size_t)mm * N + n] = c[r] + addvec[bb * N + n];
    }
  } else {
#pragma unroll
    for (int r = 0; r < 8; ++r) {
      int mm = m0 + r + hi * 8;
      D[(size_t)mm * N + n] = c[r];
    }
  }
}

// ---------------- removal branch ----------------

__global__ void prepost_kernel(const int* __restrict__ rec, int* __restrict__ pre,
                               int* __restrict__ post) {
  int idx = blockIdx.x * 256 + threadIdx.x;
  if (idx >= MROWS) return;
  int b = idx / GS;
  int r = rec[idx];
  pre[b * GS + r] = idx - b * GS;          // inverse permutation
  post[idx] = rec[b * GS + r];             // rec[b, rec[b,g]]
}

__global__ void featR_kernel(const float* __restrict__ hQ, const float* __restrict__ hK,
                             const int* __restrict__ pre, const int* __restrict__ post,
                             const float* __restrict__ sel, float* __restrict__ featR) {
  int idx = blockIdx.x * 256 + threadIdx.x;           // bs * 200
  if (idx >= BS * 2 * HALF) return;
  int b = idx / (2 * HALF), g1 = idx % (2 * HALF);
  int g = g1 + 1;
  int p = pre[b * GS + g], q = post[b * GS + g];
  const float* Qg = hQ + ((size_t)b * GS + g) * DIM;
  const float* Qp = hQ + ((size_t)b * GS + p) * DIM;
  const float* Kg = hK + ((size_t)b * GS + g) * DIM;
  const float* Kq = hK + ((size_t)b * GS + q) * DIM;
  int i = (g1 < HALF) ? g1 : g1 - HALF;
  int slot = (g1 < HALF) ? 0 : 4;
  float* frow = featR + ((size_t)b * HALF + i) * 12;
  for (int h = 0; h < 4; ++h) {
    float d1 = 0.f, d2 = 0.f, d3 = 0.f;
    int off = h * 32;
    for (int k = 0; k < 32; ++k) {
      float qp = Qp[off + k], kg = Kg[off + k], qg = Qg[off + k], kq = Kq[off + k];
      d1 += qp * kg; d2 += qg * kq; d3 += qp * kq;
    }
    frow[slot + h] = d1 + d2 - d3;
  }
  if (g1 < HALF)
    for (int s = 0; s < 4; ++s) frow[8 + s] = sel[((size_t)b * 4 + s) * HALF + i];
}

__global__ void removal_mlp_kernel(const float* __restrict__ featR,
                                   const float* __restrict__ w1, const float* __restrict__ b1,
                                   const float* __restrict__ w2, const float* __restrict__ b2,
                                   const float* __restrict__ w3, const float* __restrict__ b3,
                                   float* __restrict__ remTab) {
  int idx = blockIdx.x * 128 + threadIdx.x;
  if (idx >= BS * HALF) return;
  const float* x = featR + (size_t)idx * 12;
  float h1[32];
  for (int n = 0; n < 32; ++n) {
    float s = b1[n];
    for (int f = 0; f < 12; ++f) s += x[f] * w1[f * 32 + n];
    h1[n] = s > 0.f ? s : 0.f;
  }
  float out = b3[0];
  for (int n = 0; n < 32; ++n) {
    float s = b2[n];
    for (int f = 0; f < 32; ++f) s += h1[f] * w2[f * 32 + n];
    s = s > 0.f ? s : 0.f;
    out += s * w3[n];
  }
  remTab[idx] = tanhf(out) * 6.0f;
}

__global__ void softmax_rem_kernel(const float* __restrict__ remTab, float* __restrict__ out) {
  __shared__ float red[128];
  int b = blockIdx.x, t = threadIdx.x;
  float v = (t < HALF) ? remTab[b * HALF + t] : -INFINITY;
  red[t] = v; __syncthreads();
  for (int s = 64; s > 0; s >>= 1) { if (t < s) red[t] = fmaxf(red[t], red[t + s]); __syncthreads(); }
  float mx = red[0]; __syncthreads();
  float e = (t < HALF) ? __expf(v - mx) : 0.f;
  red[t] = e; __syncthreads();
  for (int s = 64; s > 0; s >>= 1) { if (t < s) red[t] += red[t + s]; __syncthreads(); }
  float inv = 1.f / red[0];
  if (t < HALF) out[(size_t)b * OUTW + t] = e * inv;
}

// ---------------- reinsertion branch ----------------

__global__ void qproj_kernel(const float* __restrict__ h, const int* __restrict__ fixed_action,
                             const float* __restrict__ Wq1, const float* __restrict__ Wq2,
                             float* __restrict__ Qp1, float* __restrict__ Qp2,
                             float* __restrict__ Qd1, float* __restrict__ Qd2) {
  int idx = blockIdx.x * 128 + threadIdx.x;        // b*4*128
  if (idx >= BS * 4 * DIM) return;
  int k = idx & 127, hh = (idx >> 7) & 3, b = idx >> 9;
  int pp = 1 + fixed_action[b * 3];
  const float* hp = h + ((size_t)b * GS + pp) * DIM;
  const float* hd = h + ((size_t)b * GS + pp + HALF) * DIM;
  const float* W1 = Wq1 + (size_t)hh * DIM * DIM + k;
  const float* W2 = Wq2 + (size_t)hh * DIM * DIM + k;
  float s1p = 0.f, s2p = 0.f, s1d = 0.f, s2d = 0.f;
  for (int i2 = 0; i2 < DIM; ++i2) {
    float w1v = W1[(size_t)i2 * DIM], w2v = W2[(size_t)i2 * DIM];
    float hpv = hp[i2], hdv = hd[i2];
    s1p += hpv * w1v; s2p += hpv * w2v; s1d += hdv * w1v; s2d += hdv * w2v;
  }
  Qp1[idx] = s1p; Qp2[idx] = s2p; Qd1[idx] = s1d; Qd2[idx] = s2d;
}

// fold Wk into query: E[b,hh,i] = sum_k Wk[hh,i,k] * Q[b,hh,k]
__global__ void qeff_kernel(const float* __restrict__ Qp1, const float* __restrict__ Qp2,
                            const float* __restrict__ Qd1, const float* __restrict__ Qd2,
                            const float* __restrict__ Wk1, const float* __restrict__ Wk2,
                            float* __restrict__ Ep1, float* __restrict__ Ep2,
                            float* __restrict__ Ed1, float* __restrict__ Ed2) {
  int idx = blockIdx.x * 128 + threadIdx.x;
  if (idx >= BS * 4 * DIM) return;
  int i = idx & 127, hh = (idx >> 7) & 3, b = idx >> 9;
  const float* w1row = Wk1 + ((size_t)hh * DIM + i) * DIM;
  const float* w2row = Wk2 + ((size_t)hh * DIM + i) * DIM;
  const float* q1 = Qp1 + ((size_t)b * 4 + hh) * DIM;
  const float* q2 = Qp2 + ((size_t)b * 4 + hh) * DIM;
  const float* q3 = Qd1 + ((size_t)b * 4 + hh) * DIM;
  const float* q4 = Qd2 + ((size_t)b * 4 + hh) * DIM;
  float s1 = 0.f, s2 = 0.f, s3 = 0.f, s4 = 0.f;
  for (int k = 0; k < DIM; ++k) {
    float w1v = w1row[k], w2v = w2row[k];
    s1 += w1v * q1[k]; s2 += w2v * q2[k]; s3 += w1v * q3[k]; s4 += w2v * q4[k];
  }
  Ep1[idx] = s1; Ep2[idx] = s2; Ed1[idx] = s3; Ed2[idx] = s4;
}

// A8/B8 scores + layer-1 projections
__global__ void abproj_kernel(const float* __restrict__ h, const int* __restrict__ rec,
                              const float* __restrict__ Ep1, const float* __restrict__ Ep2,
                              const float* __restrict__ Ed1, const float* __restrict__ Ed2,
                              const float* __restrict__ i_w1, const float* __restrict__ i_b1,
                              float* __restrict__ aproj, float* __restrict__ bproj) {
  int idx = blockIdx.x * 128 + threadIdx.x;     // 25728 = 201*128
  if (idx >= MROWS) return;
  int b = idx / GS;
  const float* hg = h + (size_t)idx * DIM;
  const float* hn = h + ((size_t)b * GS + rec[idx]) * DIM;
  const float norm = 0.08838834764831845f;      // 1/sqrt(128)
  float A8[8], B8[8];
  for (int hh = 0; hh < 4; ++hh) {
    const float* e1 = Ep1 + ((size_t)b * 4 + hh) * DIM;
    const float* e2 = Ep2 + ((size_t)b * 4 + hh) * DIM;
    const float* e3 = Ed1 + ((size_t)b * 4 + hh) * DIM;
    const float* e4 = Ed2 + ((size_t)b * 4 + hh) * DIM;
    float s1 = 0.f, s2 = 0.f, s3 = 0.f, s4 = 0.f;
    for (int k = 0; k < DIM; ++k) {
      float hv = hg[k], hnv = hn[k];
      s1 += hv * e1[k]; s2 += hnv * e2[k]; s3 += hv * e3[k]; s4 += hnv * e4[k];
    }
    A8[hh] = norm * s1; A8[4 + hh] = norm * s2;
    B8[hh] = norm * s3; B8[4 + hh] = norm * s4;
  }
  float* ap = aproj + (size_t)idx * 32;
  float* bp = bproj + (size_t)idx * 32;
  for (int n = 0; n < 32; ++n) {
    float sa = 0.f, sb = 0.f;
    for (int f = 0; f < 8; ++f) { sa += A8[f] * i_w1[f * 32 + n]; sb += B8[f] * i_w1[(8 + f) * 32 + n]; }
    ap[n] = sa; bp[n] = sb + i_b1[n];
  }
}

// big fused MLP over 5.17M positions: relu(aproj[i]+bproj[j]) -> WMMA 32x32 -> relu -> dot w3 -> tanh*6
// 4 waves per block; each wave owns one 16-position tile; layer-2 weights pre-packed as per-lane
// f16 fragments in global (2KB, WGP$/L2 resident) -> two 32B loads per wave, no LDS staging.
__global__ void reins_mlp_kernel(const float* __restrict__ aproj, const float* __restrict__ bproj,
                                 const _Float16* __restrict__ w2pk, const float* __restrict__ i_b2,
                                 const float* __restrict__ i_w3, const float* __restrict__ i_b3,
                                 const unsigned char* __restrict__ mask, float* __restrict__ table) {
  const int lane = threadIdx.x & 31;
  const int wv = threadIdx.x >> 5;
  const int tile = blockIdx.x * 4 + wv;
  const int b = blockIdx.y;
  const int p0 = tile * 16;
  const int l2 = lane & 15, hi = lane >> 4;
  int pos = p0 + l2;
  if (pos >= GG) pos = 0;                      // clamp; stores are guarded
  const int i = pos / GS, j = pos % GS;
  const float* ap = aproj + ((size_t)b * GS + i) * 32 + hi * 8;
  const float* bp = bproj + ((size_t)b * GS + j) * 32 + hi * 8;

  float4 ax0 = *(const float4*)(ap);
  float4 ax1 = *(const float4*)(ap + 4);
  float4 ax2 = *(const float4*)(ap + 16);
  float4 ax3 = *(const float4*)(ap + 20);
  float4 bx0 = *(const float4*)(bp);
  float4 bx1 = *(const float4*)(bp + 4);
  float4 bx2 = *(const float4*)(bp + 16);
  float4 bx3 = *(const float4*)(bp + 20);

  v16h afrag;
  {
    float x;
    x = ax0.x + bx0.x; afrag[0]  = (_Float16)(x > 0.f ? x : 0.f);
    x = ax0.y + bx0.y; afrag[1]  = (_Float16)(x > 0.f ? x : 0.f);
    x = ax0.z + bx0.z; afrag[2]  = (_Float16)(x > 0.f ? x : 0.f);
    x = ax0.w + bx0.w; afrag[3]  = (_Float16)(x > 0.f ? x : 0.f);
    x = ax1.x + bx1.x; afrag[4]  = (_Float16)(x > 0.f ? x : 0.f);
    x = ax1.y + bx1.y; afrag[5]  = (_Float16)(x > 0.f ? x : 0.f);
    x = ax1.z + bx1.z; afrag[6]  = (_Float16)(x > 0.f ? x : 0.f);
    x = ax1.w + bx1.w; afrag[7]  = (_Float16)(x > 0.f ? x : 0.f);
    x = ax2.x + bx2.x; afrag[8]  = (_Float16)(x > 0.f ? x : 0.f);
    x = ax2.y + bx2.y; afrag[9]  = (_Float16)(x > 0.f ? x : 0.f);
    x = ax2.z + bx2.z; afrag[10] = (_Float16)(x > 0.f ? x : 0.f);
    x = ax2.w + bx2.w; afrag[11] = (_Float16)(x > 0.f ? x : 0.f);
    x = ax3.x + bx3.x; afrag[12] = (_Float16)(x > 0.f ? x : 0.f);
    x = ax3.y + bx3.y; afrag[13] = (_Float16)(x > 0.f ? x : 0.f);
    x = ax3.z + bx3.z; afrag[14] = (_Float16)(x > 0.f ? x : 0.f);
    x = ax3.w + bx3.w; afrag[15] = (_Float16)(x > 0.f ? x : 0.f);
  }

  const v16h* wp = (const v16h*)(w2pk + lane * 32);
  v16h b0 = wp[0];
  v16h b1 = wp[1];

  v8f c0 = {}, c1 = {};
  c0 = wmma16x16x32(afrag, b0, c0);
  c1 = wmma16x16x32(afrag, b1, c1);

  const float w3a = i_w3[l2],      w3b = i_w3[16 + l2];
  const float b2a = i_b2[l2],      b2b = i_b2[16 + l2];
  float part[8];
#pragma unroll
  for (int r = 0; r < 8; ++r) {
    float ha = c0[r] + b2a; ha = ha > 0.f ? ha : 0.f;
    float hb = c1[r] + b2b; hb = hb > 0.f ? hb : 0.f;
    part[r] = ha * w3a + hb * w3b;
  }
#pragma unroll
  for (int r = 0; r < 8; ++r) {
    part[r] += __shfl_xor(part[r], 1, 32);
    part[r] += __shfl_xor(part[r], 2, 32);
    part[r] += __shfl_xor(part[r], 4, 32);
    part[r] += __shfl_xor(part[r], 8, 32);
  }
  if (l2 == 0) {
    float bias3 = i_b3[0];
#pragma unroll
    for (int r = 0; r < 8; ++r) {
      int mm = p0 + r + hi * 8;
      if (mm < GG) {
        float v = tanhf(part[r] + bias3) * 6.0f;
        size_t gidx = (size_t)b * GG + mm;
        if (mask[gidx]) v = -1e20f;
        table[gidx] = v;
      }
    }
  }
}

__global__ void softmax_reins_kernel(const float* __restrict__ table, float* __restrict__ out) {
  __shared__ float red[256];
  const int b = blockIdx.x, t = threadIdx.x;
  const float* row = table + (size_t)b * GG;
  float mx = -INFINITY;
  for (int k = t; k < GG; k += 256) mx = fmaxf(mx, row[k]);
  red[t] = mx; __syncthreads();
  for (int s = 128; s > 0; s >>= 1) { if (t < s) red[t] = fmaxf(red[t], red[t + s]); __syncthreads(); }
  mx = red[0]; __syncthreads();
  float sum = 0.f;
  for (int k = t; k < GG; k += 256) sum += __expf(row[k] - mx);
  red[t] = sum; __syncthreads();
  for (int s = 128; s > 0; s >>= 1) { if (t < s) red[t] += red[t + s]; __syncthreads(); }
  float inv = 1.f / red[0];
  float* orow = out + (size_t)b * OUTW + HALF;
  for (int k = t; k < GG; k += 256) orow[k] = __expf(row[k] - mx) * inv;
}

// ---------------- launch ----------------

extern "C" void kernel_launch(void* const* d_in, const int* in_sizes, int n_in,
                              void* d_out, int out_size, void* d_ws, size_t ws_size,
                              hipStream_t stream) {
  const float* h_em  = (const float*)d_in[0];
  const float* sel   = (const float*)d_in[1];
  const float* Wn    = (const float*)d_in[2];
  const float* Wg    = (const float*)d_in[3];
  const float* WQr   = (const float*)d_in[4];
  const float* WKr   = (const float*)d_in[5];
  const float* r_w1  = (const float*)d_in[6];
  const float* r_b1  = (const float*)d_in[7];
  const float* r_w2  = (const float*)d_in[8];
  const float* r_b2  = (const float*)d_in[9];
  const float* r_w3  = (const float*)d_in[10];
  const float* r_b3  = (const float*)d_in[11];
  const float* Wq1   = (const float*)d_in[12];
  const float* Wk1   = (const float*)d_in[13];
  const float* Wq2   = (const float*)d_in[14];
  const float* Wk2   = (const float*)d_in[15];
  const float* i_w1  = (const float*)d_in[16];
  const float* i_b1  = (const float*)d_in[17];
  const float* i_w2  = (const float*)d_in[18];
  const float* i_b2  = (const float*)d_in[19];
  const float* i_w3  = (const float*)d_in[20];
  const float* i_b3  = (const float*)d_in[21];
  const int*   rec   = (const int*)d_in[22];
  const int*   fact  = (const int*)d_in[23];
  const unsigned char* mask = (const unsigned char*)d_in[24];
  float* out = (float*)d_out;

  float* ws = (float*)d_ws;
  size_t off = 0;
  auto alloc = [&](size_t n) { float* p = ws + off; off += n; return p; };
  float* gmax   = alloc(BS * DIM);
  float* gproj  = alloc(BS * DIM);
  float* h      = alloc((size_t)MROWS * DIM);
  float* hQ     = alloc((size_t)MROWS * DIM);
  float* hK     = alloc((size_t)MROWS * DIM);
  _Float16* Wn_pk  = (_Float16*)alloc(8192);   // 16384 halves
  _Float16* WQr_pk = (_Float16*)alloc(8192);
  _Float16* WKr_pk = (_Float16*)alloc(8192);
  _Float16* w2pk   = (_Float16*)alloc(512);    // 1024 halves
  float* featR  = alloc((size_t)BS * HALF * 12);
  float* remTab = alloc(BS * HALF);
  float* Qp1 = alloc(BS * 4 * DIM); float* Qp2 = alloc(BS * 4 * DIM);
  float* Qd1 = alloc(BS * 4 * DIM); float* Qd2 = alloc(BS * 4 * DIM);
  float* Ep1 = alloc(BS * 4 * DIM); float* Ep2 = alloc(BS * 4 * DIM);
  float* Ed1 = alloc(BS * 4 * DIM); float* Ed2 = alloc(BS * 4 * DIM);
  float* aproj = alloc((size_t)MROWS * 32);
  float* bproj = alloc((size_t)MROWS * 32);
  float* table = alloc((size_t)BS * GG);
  int* pre  = (int*)alloc(MROWS);
  int* post = (int*)alloc(MROWS);

  // h = h_em@Wn + broadcast(gmax@Wg)
  gmax_kernel <<<BS, DIM, 0, stream>>>(h_em, gmax);
  gproj_kernel<<<BS, DIM, 0, stream>>>(gmax, Wg, gproj);
  pack_bfrag_kernel<<<64, 256, 0, stream>>>(Wn,  Wn_pk,  0);
  pack_bfrag_kernel<<<64, 256, 0, stream>>>(WQr, WQr_pk, 1);
  pack_bfrag_kernel<<<64, 256, 0, stream>>>(WKr, WKr_pk, 1);
  pack_w2_kernel<<<1, 256, 0, stream>>>(i_w2, w2pk);

  dim3 gemmGrid(MROWS / 16, DIM / 16);
  gemm_wmma<<<gemmGrid, 32, 0, stream>>>(h_em, Wn_pk,  h,  gproj);
  gemm_wmma<<<gemmGrid, 32, 0, stream>>>(h,    WQr_pk, hQ, nullptr);
  gemm_wmma<<<gemmGrid, 32, 0, stream>>>(h,    WKr_pk, hK, nullptr);

  // removal branch
  prepost_kernel<<<(MROWS + 255) / 256, 256, 0, stream>>>(rec, pre, post);
  featR_kernel<<<(BS * 2 * HALF + 255) / 256, 256, 0, stream>>>(hQ, hK, pre, post, sel, featR);
  removal_mlp_kernel<<<(BS * HALF + 127) / 128, 128, 0, stream>>>(featR, r_w1, r_b1, r_w2, r_b2, r_w3, r_b3, remTab);
  softmax_rem_kernel<<<BS, 128, 0, stream>>>(remTab, out);

  // reinsertion branch
  qproj_kernel<<<(BS * 4 * DIM + 127) / 128, 128, 0, stream>>>(h, fact, Wq1, Wq2, Qp1, Qp2, Qd1, Qd2);
  qeff_kernel <<<(BS * 4 * DIM + 127) / 128, 128, 0, stream>>>(Qp1, Qp2, Qd1, Qd2, Wk1, Wk2, Ep1, Ep2, Ed1, Ed2);
  abproj_kernel<<<MROWS / 128, 128, 0, stream>>>(h, rec, Ep1, Ep2, Ed1, Ed2, i_w1, i_b1, aproj, bproj);

  const int ntiles = (GG + 15) / 16;           // 2526
  dim3 mlpGrid((ntiles + 3) / 4, BS);
  reins_mlp_kernel<<<mlpGrid, 128, 0, stream>>>(aproj, bproj, w2pk, i_b2, i_w3, i_b3, mask, table);
  softmax_reins_kernel<<<BS, 256, 0, stream>>>(table, out);
}
